// NeuromodulatedAttentionWithElectrodes_2508260901588
// MI455X (gfx1250) — compile-verified, matrix operand order checked by reading, not run on hardware
//
#include <hip/hip_runtime.h>
#include <hip/hip_bf16.h>

// MI455X / gfx1250 fused neuromodulated-attention.
// Key optimization: the neuromodulation term (dopamine - serotonin) is a
// per-row constant added to `scores` *before* row standardization; it cancels
// exactly in (ms - mean) and does not change the (ddof=1) std, so the whole
// MLP branch is dead math. Output depends only on
//   fc( softmax( standardize(QK^T / sqrt(D)) ) @ V ) + fc_b.
// Matmuls run on v_wmma_f32_16x16x32_bf16 (bf16 in, f32 accumulate).

typedef __attribute__((ext_vector_type(16))) __bf16 v16bf;
typedef __attribute__((ext_vector_type(8)))  float  v8f;

#define B_SZ   64
#define N_SZ   512
#define D_SZ   1024
#define C_SZ   4
#define TILE   16
#define NWAVE  8
#define THREADS 256

union BfVec {
  v16bf v;
  unsigned short s[16];
  uint4 u4[2];
};

__device__ __forceinline__ unsigned short f2bf(float f) {
  __bf16 h = (__bf16)f;
  return __builtin_bit_cast(unsigned short, h);
}

__device__ __forceinline__ float wredsum(float x) {
#pragma unroll
  for (int m = 16; m >= 1; m >>= 1) x += __shfl_xor(x, m, 32);
  return x;
}
__device__ __forceinline__ float wredmax(float x) {
#pragma unroll
  for (int m = 16; m >= 1; m >>= 1) x = fmaxf(x, __shfl_xor(x, m, 32));
  return x;
}

// Dynamic LDS layout (bytes):
//   sQ  : bf16[16][1024]   @ 0       (32768)
//   sS  : f32 [16][512]    @ 32768   (32768)
//   sP  : bf16[16][512]    @ 65536   (16384)
//   sVt : bf16[1024][32]   @ 81920   (65536)   (V^T chunk: [d][k])
//   sWr : f32 [8][4]       @ 147456  (128)
#define SMEM_BYTES (147456 + 128)

__global__ void __launch_bounds__(THREADS)
neuromod_attn_kernel(const float* __restrict__ Q, const float* __restrict__ K,
                     const float* __restrict__ V, const int* __restrict__ ids,
                     const float* __restrict__ emb, const float* __restrict__ fcw,
                     float* __restrict__ ws) {
  extern __shared__ char smem[];
  unsigned short* sQ  = (unsigned short*)(smem);           // bf16[16][1024]
  float*          sS  = (float*)(smem + 32768);            // [16][512]
  unsigned short* sP  = (unsigned short*)(smem + 65536);   // bf16[16][512]
  unsigned short* sVt = (unsigned short*)(smem + 81920);   // bf16[1024][32]
  float*          sWr = (float*)(smem + 147456);           // [8][4]

  const int tid   = threadIdx.x;
  const int wave  = tid >> 5;
  const int lane  = tid & 31;
  const int lhalf = lane >> 4;   // selects k sub-chunk (0 or 1)
  const int l15   = lane & 15;
  const int rowTile = blockIdx.x;          // 0..31
  const int b       = blockIdx.y;          // 0..63
  const int rowbase = rowTile * TILE;

  const v8f zero8 = {0.f, 0.f, 0.f, 0.f, 0.f, 0.f, 0.f, 0.f};

  // ---- Phase 1: stage Qmod = Q + emb[ids] as bf16 in LDS -------------------
  for (int i = tid; i < TILE * D_SZ / 4; i += THREADS) {
    const int r = (i * 4) / D_SZ, c = (i * 4) % D_SZ;
    const int er = ids[rowbase + r];
    const float4 q4 = *(const float4*)(Q + ((size_t)b * N_SZ + rowbase + r) * D_SZ + c);
    const float4 e4 = *(const float4*)(emb + (size_t)er * D_SZ + c);
    uint2 p;
    p.x = (unsigned)f2bf(q4.x + e4.x) | ((unsigned)f2bf(q4.y + e4.y) << 16);
    p.y = (unsigned)f2bf(q4.z + e4.z) | ((unsigned)f2bf(q4.w + e4.w) << 16);
    *(uint2*)(sQ + r * D_SZ + c) = p;
  }
  __syncthreads();

  // ---- Phase 2: scores[16][512] = Qmod @ Kmod^T / sqrt(D) ------------------
  // wave w owns score columns n in [64w, 64w+64): 4 WMMA n-tiles.
  {
    v8f acc[4];
#pragma unroll
    for (int j = 0; j < 4; ++j) acc[j] = zero8;

    const int nbase = wave * 64;
    for (int kt = 0; kt < D_SZ / 32; ++kt) {
      const int kc = kt * 32 + lhalf * 16;
      BfVec a;  // A operand: Q row l15, 16 contiguous bf16 at kc
      const uint4* qa = (const uint4*)(sQ + l15 * D_SZ + kc);
      a.u4[0] = qa[0];
      a.u4[1] = qa[1];
#pragma unroll
      for (int j = 0; j < 4; ++j) {
        const int kr = nbase + j * 16 + l15;   // K electrode row (score column)
        const int er = ids[kr];
        const float* kp = K + ((size_t)b * N_SZ + kr) * D_SZ + kc;
        const float* ep = emb + (size_t)er * D_SZ + kc;
        BfVec bb;
#pragma unroll
        for (int t = 0; t < 4; ++t) {
          const float4 kv = ((const float4*)kp)[t];
          const float4 ev = ((const float4*)ep)[t];
          bb.s[4 * t + 0] = f2bf(kv.x + ev.x);
          bb.s[4 * t + 1] = f2bf(kv.y + ev.y);
          bb.s[4 * t + 2] = f2bf(kv.z + ev.z);
          bb.s[4 * t + 3] = f2bf(kv.w + ev.w);
        }
        acc[j] = __builtin_amdgcn_wmma_f32_16x16x32_bf16(
            false, a.v, false, bb.v, (short)0, acc[j], false, false);
      }
      if (kt + 1 < D_SZ / 32) {  // gfx1250 global_prefetch_b8 for next K chunk
        const int kr0 = nbase + l15;
        __builtin_prefetch(K + ((size_t)b * N_SZ + kr0) * D_SZ + (kt + 1) * 32 + lhalf * 16, 0, 1);
      }
    }
    const float inv_sqrt_d = 0.03125f;  // 1/sqrt(1024)
#pragma unroll
    for (int j = 0; j < 4; ++j) {
      const int n = nbase + j * 16 + l15;
#pragma unroll
      for (int r = 0; r < 8; ++r) {
        const int m = r + 8 * lhalf;
        sS[m * N_SZ + n] = acc[j][r] * inv_sqrt_d;
      }
    }
  }
  __syncthreads();

  // ---- Phase 3: per-row standardize (ddof=1) + softmax -> probs2 (bf16) ----
#pragma unroll
  for (int mi = 0; mi < 2; ++mi) {
    const int m = wave * 2 + mi;
    float x[16];
    float s = 0.f, ss2 = 0.f;
#pragma unroll
    for (int i = 0; i < 16; ++i) {
      x[i] = sS[m * N_SZ + lane + 32 * i];
      s += x[i];
      ss2 += x[i] * x[i];
    }
    s = wredsum(s);
    ss2 = wredsum(ss2);
    const float mean = s * (1.f / 512.f);
    float var = (ss2 - 512.f * mean * mean) * (1.f / 511.f);
    var = fmaxf(var, 0.f);
    const float inv = 1.f / (sqrtf(var) + 1e-6f);
    float mx = -3.4e38f;
#pragma unroll
    for (int i = 0; i < 16; ++i) {
      x[i] = (x[i] - mean) * inv;
      mx = fmaxf(mx, x[i]);
    }
    mx = wredmax(mx);
    float se = 0.f;
#pragma unroll
    for (int i = 0; i < 16; ++i) {
      x[i] = __expf(x[i] - mx);
      se += x[i];
    }
    se = wredsum(se);
    const float rse = 1.f / se;
#pragma unroll
    for (int i = 0; i < 16; ++i) sP[m * N_SZ + lane + 32 * i] = f2bf(x[i] * rse);
  }
  __syncthreads();

  // ---- Phase 4: attn = probs2 @ Vmod, fused with fc partial dot ------------
  // wave w owns d-tiles [8w, 8w+8); V chunks staged transposed in LDS.
  v8f accd[8];
#pragma unroll
  for (int j = 0; j < 8; ++j) accd[j] = zero8;
  float pc[C_SZ] = {0.f, 0.f, 0.f, 0.f};

  for (int kt = 0; kt < N_SZ / 32; ++kt) {
    const int k0 = kt * 32;
    for (int i = tid; i < 32 * D_SZ / 4; i += THREADS) {
      const int r = (i * 4) / D_SZ, c = (i * 4) % D_SZ;
      const int er = ids[k0 + r];
      const float4 v4 = *(const float4*)(V + ((size_t)b * N_SZ + k0 + r) * D_SZ + c);
      const float4 e4 = *(const float4*)(emb + (size_t)er * D_SZ + c);
      sVt[(c + 0) * 32 + r] = f2bf(v4.x + e4.x);
      sVt[(c + 1) * 32 + r] = f2bf(v4.y + e4.y);
      sVt[(c + 2) * 32 + r] = f2bf(v4.z + e4.z);
      sVt[(c + 3) * 32 + r] = f2bf(v4.w + e4.w);
    }
    __syncthreads();
    BfVec a;  // A = probs2 row l15, k chunk
    const uint4* pa = (const uint4*)(sP + l15 * N_SZ + k0 + lhalf * 16);
    a.u4[0] = pa[0];
    a.u4[1] = pa[1];
#pragma unroll
    for (int j = 0; j < 8; ++j) {
      const int d = (wave * 8 + j) * 16 + l15;
      BfVec bb;  // B = Vmod^T column d, 16 contiguous k values
      const uint4* vb = (const uint4*)(sVt + d * 32 + lhalf * 16);
      bb.u4[0] = vb[0];
      bb.u4[1] = vb[1];
      accd[j] = __builtin_amdgcn_wmma_f32_16x16x32_bf16(
          false, a.v, false, bb.v, (short)0, accd[j], false, false);
    }
    __syncthreads();
  }

  // fc partials: pc[c] += attn[m][d] * fc_w[c, (rowbase+m)*D + d]
#pragma unroll
  for (int j = 0; j < 8; ++j) {
    const int d = (wave * 8 + j) * 16 + l15;
#pragma unroll
    for (int r = 0; r < 8; ++r) {
      const int m = r + 8 * lhalf;
      const float av = accd[j][r];
      const size_t base = (size_t)(rowbase + m) * D_SZ + d;
#pragma unroll
      for (int c = 0; c < C_SZ; ++c)
        pc[c] += av * fcw[(size_t)c * ((size_t)N_SZ * D_SZ) + base];
    }
  }
#pragma unroll
  for (int c = 0; c < C_SZ; ++c) {
    const float v = wredsum(pc[c]);
    if (lane == 0) sWr[wave * C_SZ + c] = v;
  }
  __syncthreads();
  if (tid < C_SZ) {  // deterministic fixed-order cross-wave reduce
    float t = 0.f;
#pragma unroll
    for (int w = 0; w < NWAVE; ++w) t += sWr[w * C_SZ + tid];
    ws[((size_t)b * 32 + rowTile) * C_SZ + tid] = t;
  }
}

__global__ void __launch_bounds__(B_SZ * C_SZ)
finish_kernel(const float* __restrict__ ws, const float* __restrict__ fcb,
              float* __restrict__ out) {
  const int t = threadIdx.x;  // 256 = B*C
  const int b = t >> 2, c = t & 3;
  float s = fcb[c];
#pragma unroll
  for (int i = 0; i < 32; ++i) s += ws[((size_t)b * 32 + i) * C_SZ + c];
  out[t] = s;
}

extern "C" void kernel_launch(void* const* d_in, const int* in_sizes, int n_in,
                              void* d_out, int out_size, void* d_ws, size_t ws_size,
                              hipStream_t stream) {
  (void)in_sizes; (void)n_in; (void)out_size; (void)ws_size;
  const float* Q   = (const float*)d_in[0];
  const float* K   = (const float*)d_in[1];
  const float* V   = (const float*)d_in[2];
  const int*   ids = (const int*)d_in[3];
  const float* emb = (const float*)d_in[4];
  // d_in[5..14]: dopamine/serotonin MLP params + scales — provably no-ops
  // (row-constant shift cancels in the pre-softmax standardization).
  const float* fcw = (const float*)d_in[15];
  const float* fcb = (const float*)d_in[16];
  float* out = (float*)d_out;
  float* ws  = (float*)d_ws;

  hipFuncSetAttribute((const void*)neuromod_attn_kernel,
                      hipFuncAttributeMaxDynamicSharedMemorySize, SMEM_BYTES);
  dim3 grid(N_SZ / TILE, B_SZ);
  neuromod_attn_kernel<<<grid, THREADS, SMEM_BYTES, stream>>>(Q, K, V, ids, emb, fcw, ws);
  finish_kernel<<<1, B_SZ * C_SZ, 0, stream>>>(ws, fcb, out);
}
